// EncoderRNN_12128987644544
// MI455X (gfx1250) — matrix operand order repaired and checked by reading, not compiled, not run on hardware
//
#include <hip/hip_runtime.h>

// ---------------------------------------------------------------------------
// GRU (2 layers x 2 forward-in-time directions), B=64, T=512, I=H=256.
// One persistent 256-thread workgroup (8 waves) per (layer,dir) cell.
//  - all GEMMs: v_wmma_f32_16x16x32_bf16, fp32 accumulate
//  - hidden state: fp32 plane in LDS (owner-wave exclusive cells) +
//    bf16 mirror in LDS as the WMMA A operand
//  - x_t staged into LDS by the Tensor Data Mover (double-buffered, prefetch
//    t+1 during compute of t), synced via s_wait_tensorcnt + barrier
//  - weights + input pre-converted to bf16 once (L2-resident afterwards)
//  - tile loop kept ROLLED (O(1-tile) register state) to avoid spills
// ---------------------------------------------------------------------------

typedef __attribute__((ext_vector_type(16))) __bf16 v16bf;
typedef __attribute__((ext_vector_type(8)))  __bf16 v8bf;
typedef __attribute__((ext_vector_type(4)))  __bf16 v4bf;
typedef __attribute__((ext_vector_type(8)))  float  v8f;
typedef unsigned int v4u __attribute__((ext_vector_type(4)));
typedef int          v8i __attribute__((ext_vector_type(8)));
typedef int          v4i __attribute__((ext_vector_type(4)));

#define B_  64
#define T_  512
#define H_  256
#define G3_ 768   // 3*H, gate order r,z,n

#if __has_builtin(__builtin_amdgcn_tensor_load_to_lds) && \
    __has_builtin(__builtin_amdgcn_s_wait_tensorcnt)
#define USE_TDM 1
#else
#define USE_TDM 0
#endif

__device__ __forceinline__ v16bf frag_join(v8bf lo, v8bf hi) {
  union { v16bf v; v8bf h[2]; } u;
  u.h[0] = lo; u.h[1] = hi;
  return u.v;
}
// A fragment (16x32): lane m=lane&15 holds k = k0..k0+7 and k0+16..k0+23
__device__ __forceinline__ v16bf a_frag_bf(const __bf16* row, int k0) {
  return frag_join(*(const v8bf*)(row + k0), *(const v8bf*)(row + k0 + 16));
}
// B fragment (32x16): lane n=lane&15 holds k = k0..k0+7 and k0+16..k0+23 of W row n
__device__ __forceinline__ v16bf b_frag_bf(const __bf16* row, int k0) {
  return frag_join(*(const v8bf*)(row + k0), *(const v8bf*)(row + k0 + 16));
}
__device__ __forceinline__ v8f wmma_bf16(v16bf a, v16bf b, v8f c) {
  return __builtin_amdgcn_wmma_f32_16x16x32_bf16(false, a, false, b,
                                                 (short)0, c, false, false);
}
__device__ __forceinline__ float sigmoidf_(float x) {
  return 1.f / (1.f + __expf(-x));
}

__global__ void cvt_bf16_kernel(const float* __restrict__ s,
                                __bf16* __restrict__ d, int n) {
  int i = blockIdx.x * blockDim.x + threadIdx.x;
  if (i < n) d[i] = (__bf16)s[i];
}

#if USE_TDM
typedef __attribute__((address_space(3))) char lds_char;
__device__ __forceinline__ unsigned lds_off_of(const void* p) {
  return (unsigned)(uintptr_t)(lds_char*)p;
}
// 2D TDM: tile = 64 rows x kx bf16, row stride T_*kx elements.
__device__ __forceinline__ void tdm_load_tile(const __bf16* g, unsigned ldsoff,
                                              int kx) {
  unsigned long long ga = (unsigned long long)(uintptr_t)g;
  v4u g0;
  g0[0] = 1u;                                        // count=1
  g0[1] = ldsoff;                                    // lds_addr (bytes)
  g0[2] = (unsigned)ga;                              // global_addr lo
  g0[3] = (unsigned)((ga >> 32) & 0x01FFFFFFu) | 0x80000000u; // hi | type=2
  v8i g1;
  g1[0] = 0x00010000;               // data_size=1 (2 bytes/elem)
  g1[1] = (kx & 0xFFFF) << 16;      // tensor_dim0[15:0]
  g1[2] = (64 << 16);               // tensor_dim0 hi=0 | tensor_dim1 lo=64
  g1[3] = (kx << 16);               // tensor_dim1 hi=0 | tile_dim0=kx
  g1[4] = 64;                       // tile_dim1=64, tile_dim2=0
  unsigned long long st = (unsigned long long)T_ * (unsigned)kx;
  g1[5] = (int)(unsigned)st;        // tensor_dim0_stride lo
  g1[6] = (int)((st >> 32) & 0xFFFFu); // stride hi | dim1_stride lo=0
  g1[7] = 0;
  v4i gz4; gz4[0] = 0; gz4[1] = 0; gz4[2] = 0; gz4[3] = 0;
  v8i gz8; gz8[0] = 0; gz8[1] = 0; gz8[2] = 0; gz8[3] = 0;
           gz8[4] = 0; gz8[5] = 0; gz8[6] = 0; gz8[7] = 0;
  __builtin_amdgcn_tensor_load_to_lds(g0, g1, gz4, gz4, gz8, 0);
}
#endif

// One block per direction, 256 threads = 8 waves.
// Wave w: M-tile = w>>1 (batch rows), h-column tiles ((w&1)*8 + i)*16, i=0..7.
template <int LAYER>
__global__ __launch_bounds__(256)
void gru_layer_kernel(const __bf16* __restrict__ xin_bf, // (B,T,KX) bf16
                      const __bf16* __restrict__ Wi,     // [2][768][KX] bf16
                      const __bf16* __restrict__ Wh,     // [2][768][256] bf16
                      const float*  __restrict__ b_ih,   // [2][768]
                      const float*  __restrict__ b_hh,   // [2][768]
                      const int*    __restrict__ seq_len,
                      __bf16*       __restrict__ out_bf, // layer0: (B,T,2H)
                      float*        __restrict__ dout)   // layer1: (B,2H)
{
  constexpr int KX = (LAYER == 0) ? 256 : 512;
  extern __shared__ __align__(16) char smem[];
  __bf16* hsm  = (__bf16*)smem;                       // 64x256 bf16 (32 KB)
  float*  hf32 = (float*)(smem + 32768);              // 64x256 fp32 (64 KB)
  float*  bias_r  = (float*)(smem + 32768 + 65536);   // 4 x 256 fp32 (4 KB)
  float*  bias_z  = bias_r + H_;
  float*  bias_xn = bias_z + H_;
  float*  bias_hn = bias_xn + H_;
  __bf16* xb0 = (__bf16*)(smem + 32768 + 65536 + 4096); // 64xKX bf16
  __bf16* xb1 = xb0 + B_ * KX;                          // double buffer

  const int dir  = blockIdx.x;
  const int tid  = threadIdx.x;
  const int wave = tid >> 5, lane = tid & 31, lh = lane >> 4, ln = lane & 15;
  const int mtile = wave >> 1;
  const int nbase = (wave & 1) * 8;

  const __bf16* Wid = Wi + (size_t)dir * G3_ * KX;
  const __bf16* Whd = Wh + (size_t)dir * G3_ * H_;
  const float*  bi  = b_ih + dir * G3_;
  const float*  bh  = b_hh + dir * G3_;

  // init LDS: zero hidden state, fill bias tables
  for (int i = tid; i < B_ * H_; i += 256) hsm[i] = (__bf16)0.f;
  for (int i = tid; i < B_ * H_; i += 256) hf32[i] = 0.f;
  for (int c = tid; c < H_; c += 256) {
    bias_r[c]  = bi[c] + bh[c];
    bias_z[c]  = bi[H_ + c] + bh[H_ + c];
    bias_xn[c] = bi[2 * H_ + c];
    bias_hn[c] = bh[2 * H_ + c];
  }

  int sl[8];
  if constexpr (LAYER == 1) {
#pragma unroll
    for (int e = 0; e < 8; ++e) sl[e] = seq_len[mtile * 16 + lh * 8 + e];
  }

#if USE_TDM
  if (wave == 0) tdm_load_tile(xin_bf, lds_off_of(xb0), KX);
#endif
  __syncthreads();

  int cur = 0;
#pragma unroll 1
  for (int t = 0; t < T_; ++t) {
    const __bf16* xb;
#if USE_TDM
    if (wave == 0) __builtin_amdgcn_s_wait_tensorcnt(0);
    __syncthreads();                      // x tile for t visible to all
    if (wave == 0 && t + 1 < T_)          // prefetch t+1 into other buffer
      tdm_load_tile(xin_bf + (size_t)(t + 1) * KX,
                    lds_off_of(cur ? xb0 : xb1), KX);
    xb = cur ? xb1 : xb0;
#else
    {
      const int CH = KX / 8;
      for (int idx = tid; idx < B_ * CH; idx += 256) {
        int row = idx / CH, c0 = (idx % CH) * 8;
        *(uint4*)&xb0[row * KX + c0] =
            *(const uint4*)&xin_bf[((size_t)row * T_ + t) * KX + c0];
      }
      __syncthreads();
    }
    xb = xb0;
#endif

    const __bf16* xrow = xb + (mtile * 16 + ln) * KX;
    const __bf16* hrow = hsm + (mtile * 16 + ln) * H_;

#pragma unroll 1                    // rolled: O(1-tile) register state
    for (int i = 0; i < 8; ++i) {
      const int ncol = (nbase + i) * 16;
      const int g    = ncol + ln;

      v8f accr, accz, accxn, acchn;
#pragma unroll
      for (int e = 0; e < 8; ++e) {
        accr[e] = 0.f; accz[e] = 0.f; accxn[e] = 0.f; acchn[e] = 0.f;
      }

      // ---- gx from LDS-staged x tile (r, z, n_x) ----
      const __bf16* wr = Wid + (size_t)g * KX;
      const __bf16* wz = Wid + (size_t)(H_ + g) * KX;
      const __bf16* wn = Wid + (size_t)(2 * H_ + g) * KX;
      for (int kt = 0; kt < KX / 32; ++kt) {
        const int ka = kt * 32 + lh * 8, kb = kt * 32 + lh * 8;
        v16bf a  = a_frag_bf(xrow, ka);
        v16bf br = b_frag_bf(wr, kb);
        v16bf bz = b_frag_bf(wz, kb);
        v16bf bn = b_frag_bf(wn, kb);
        accr  = wmma_bf16(a, br, accr);
        accz  = wmma_bf16(a, bz, accz);
        accxn = wmma_bf16(a, bn, accxn);
      }

      // ---- gh from LDS hidden state (r, z, n_h) ----
      const __bf16* vr = Whd + (size_t)g * H_;
      const __bf16* vz = Whd + (size_t)(H_ + g) * H_;
      const __bf16* vn = Whd + (size_t)(2 * H_ + g) * H_;
      for (int kt = 0; kt < H_ / 32; ++kt) {
        const int ka = kt * 32 + lh * 8, kb = kt * 32 + lh * 8;
        v16bf a  = a_frag_bf(hrow, ka);
        v16bf br = b_frag_bf(vr, kb);
        v16bf bz = b_frag_bf(vz, kb);
        v16bf bn = b_frag_bf(vn, kb);
        accr  = wmma_bf16(a, br, accr);
        accz  = wmma_bf16(a, bz, accz);
        acchn = wmma_bf16(a, bn, acchn);
      }

      // ---- gates + fp32 state update; hf32 cells are owner-exclusive ----
      const int col = ncol + ln;
      const float br_ = bias_r[col], bz_ = bias_z[col];
      const float bx_ = bias_xn[col], bn_ = bias_hn[col];
#pragma unroll
      for (int e = 0; e < 8; ++e) {
        const int m = mtile * 16 + lh * 8 + e;
        float r = sigmoidf_(accr[e] + br_);
        float z = sigmoidf_(accz[e] + bz_);
        float n = tanhf((accxn[e] + bx_) + r * (acchn[e] + bn_));
        float h = (1.f - z) * n + z * hf32[m * H_ + col];
        hf32[m * H_ + col] = h;
        if constexpr (LAYER == 1) {
          if (t == sl[e] - 1)
            dout[(size_t)m * (2 * H_) + dir * H_ + col] = h;
        }
      }
    }

    __syncthreads();   // all reads of old hsm + all hf32 updates done

    // fused publish: hf32 -> hsm (bf16), and for layer0 also -> global out
    for (int idx = tid * 4; idx < B_ * H_; idx += 256 * 4) {
      float4 v = *(const float4*)&hf32[idx];
      v4bf b;
      b[0] = (__bf16)v.x; b[1] = (__bf16)v.y;
      b[2] = (__bf16)v.z; b[3] = (__bf16)v.w;
      *(v4bf*)&hsm[idx] = b;
      if constexpr (LAYER == 0) {
        const int row = idx >> 8, c0 = idx & 255;
        *(v4bf*)&out_bf[((size_t)row * T_ + t) * (2 * H_) + dir * H_ + c0] = b;
      }
    }
    __syncthreads();   // new hsm published
    cur ^= 1;
  }
}

extern "C" void kernel_launch(void* const* d_in, const int* in_sizes, int n_in,
                              void* d_out, int out_size, void* d_ws, size_t ws_size,
                              hipStream_t stream) {
  (void)in_sizes; (void)n_in; (void)out_size; (void)ws_size;

  const float* x    = (const float*)d_in[0];
  const int*   slen = (const int*)d_in[1];
  const float* Wih0 = (const float*)d_in[2];
  const float* Whh0 = (const float*)d_in[3];
  const float* bih0 = (const float*)d_in[4];
  const float* bhh0 = (const float*)d_in[5];
  const float* Wih1 = (const float*)d_in[6];
  const float* Whh1 = (const float*)d_in[7];
  const float* bih1 = (const float*)d_in[8];
  const float* bhh1 = (const float*)d_in[9];
  float* out = (float*)d_out;

  // workspace (bf16 elements), ~54 MB total
  __bf16* xbf = (__bf16*)d_ws;                  // 64*512*256
  __bf16* Wi0 = xbf + (size_t)B_ * T_ * 256;    // 2*768*256
  __bf16* Wh0 = Wi0 + 2 * 768 * 256;            // 2*768*256
  __bf16* Wi1 = Wh0 + 2 * 768 * 256;            // 2*768*512
  __bf16* Wh1 = Wi1 + 2 * 768 * 512;            // 2*768*256
  __bf16* o0  = Wh1 + 2 * 768 * 256;            // 64*512*512

  auto cvt = [&](const float* s, __bf16* d, int n) {
    cvt_bf16_kernel<<<(n + 255) / 256, 256, 0, stream>>>(s, d, n);
  };
  cvt(x,    xbf, B_ * T_ * 256);
  cvt(Wih0, Wi0, 2 * 768 * 256);
  cvt(Whh0, Wh0, 2 * 768 * 256);
  cvt(Wih1, Wi1, 2 * 768 * 512);
  cvt(Whh1, Wh1, 2 * 768 * 256);

  const size_t base  = 32768 + 65536 + 4096;           // hsm + hf32 + bias
  const size_t smem0 = base + 2 * (size_t)B_ * 256 * 2; // ~164 KB
  const size_t smem1 = base + 2 * (size_t)B_ * 512 * 2; // ~228 KB
  (void)hipFuncSetAttribute((const void*)&gru_layer_kernel<0>,
                            hipFuncAttributeMaxDynamicSharedMemorySize, (int)smem0);
  (void)hipFuncSetAttribute((const void*)&gru_layer_kernel<1>,
                            hipFuncAttributeMaxDynamicSharedMemorySize, (int)smem1);

  gru_layer_kernel<0><<<2, 256, smem0, stream>>>(xbf, Wi0, Wh0, bih0, bhh0,
                                                 nullptr, o0, nullptr);
  gru_layer_kernel<1><<<2, 256, smem1, stream>>>(o0, Wi1, Wh1, bih1, bhh1,
                                                 slen, nullptr, out);
}
